// GraphGenerator_89292370084346
// MI455X (gfx1250) — matrix operand order repaired and checked by reading, not compiled, not run on hardware
//
#include <hip/hip_runtime.h>

// ---------------------------------------------------------------------------
// sim = Z Z^T with Z[n] = concat(wn_0[n], wn_1[n]) / sqrt(2)  (f16, K=512)
// then threshold (<0.2 -> 0) fused into GEMM epilogue, then per-row top-K mask.
// GEMM in v_wmma_f32_16x16x32_f16; A/B tiles staged to LDS via gfx1250 async
// global->LDS copies (ASYNCcnt) when the builtin exists.
// ---------------------------------------------------------------------------

typedef _Float16 v16h __attribute__((ext_vector_type(16)));
typedef _Float16 v8h  __attribute__((ext_vector_type(8)));
typedef float    v8f  __attribute__((ext_vector_type(8)));
typedef int      v4i  __attribute__((ext_vector_type(4)));

typedef __attribute__((address_space(1))) v4i* gp4;   // global 16B unit
typedef __attribute__((address_space(3))) v4i* lp4;   // LDS 16B unit

#define N_ROWS  8192
#define DIM     256
#define KDIM    512        // NHEAD * DIM
#define THRESH  0.2f

// ---- gfx1250 async global->LDS copy (guarded; sync fallback) --------------
#if defined(__has_builtin)
#if __has_builtin(__builtin_amdgcn_global_load_async_to_lds_b128)
#define HAVE_ASYNC_LDS 1
#endif
#if __has_builtin(__builtin_amdgcn_s_wait_asynccnt)
#define HAVE_ASYNC_WAIT 1
#endif
#endif

__device__ __forceinline__ void async_cp16(const _Float16* g, _Float16* l) {
#ifdef HAVE_ASYNC_LDS
  __builtin_amdgcn_global_load_async_to_lds_b128(
      (gp4)(void*)g, (lp4)(void*)l, 0, 0);
#else
  *(v8h*)l = *(const v8h*)g;   // synchronous fallback
#endif
}

__device__ __forceinline__ void async_wait_all() {
#ifdef HAVE_ASYNC_LDS
#ifdef HAVE_ASYNC_WAIT
  __builtin_amdgcn_s_wait_asynccnt(0);
#else
  asm volatile("s_wait_asynccnt 0" ::: "memory");
#endif
#endif
}

// ---------------------------------------------------------------------------
// Kernel 1: build Z [8192 x 512] f16.  One wave (32 lanes) per row; each lane
// owns 8 of the 256 columns; cross-lane reduction for the per-head norm.
// z = (x*w_h) / max(||x*w_h||, 1e-8) / sqrt(2)
// ---------------------------------------------------------------------------
__global__ __launch_bounds__(256) void build_z_kernel(
    const float* __restrict__ x, const float* __restrict__ w,
    _Float16* __restrict__ Z) {
  const int lane = threadIdx.x & 31;
  const int wv   = threadIdx.x >> 5;
  const int row  = blockIdx.x * 8 + wv;

  float xv[8];
  const float* xr = x + (size_t)row * DIM + lane * 8;
#pragma unroll
  for (int i = 0; i < 8; ++i) xv[i] = xr[i];

#pragma unroll
  for (int h = 0; h < 2; ++h) {
    const float* wr = w + h * DIM + lane * 8;
    float p[8];
    float ss = 0.0f;
#pragma unroll
    for (int i = 0; i < 8; ++i) { p[i] = xv[i] * wr[i]; ss += p[i] * p[i]; }
    // wave32 reduction
#pragma unroll
    for (int off = 16; off > 0; off >>= 1) ss += __shfl_xor(ss, off, 32);
    const float nrm   = sqrtf(ss);
    const float scale = 0.70710678118f / fmaxf(nrm, 1e-8f); // 1/(max(||.||,eps)*sqrt(H))
    v8h z;
#pragma unroll
    for (int i = 0; i < 8; ++i) z[i] = (_Float16)(p[i] * scale);
    *(v8h*)(Z + (size_t)row * KDIM + h * DIM + lane * 8) = z;
  }
}

// ---------------------------------------------------------------------------
// Kernel 2: sim = Z Z^T, thresholded.  Block tile 128x64 (8 waves, each wave
// a 32x32 sub-tile = 2x2 WMMA fragments).  K staged in 4 chunks of 128 halves.
// LDS row stride padded to 136 halves (272 B) to avoid bank conflicts.
// ---------------------------------------------------------------------------
#define BM  128
#define BN  64
#define KC  128
#define LDA 136

__global__ __launch_bounds__(256) void gemm_sim_kernel(
    const _Float16* __restrict__ Z, float* __restrict__ out) {
  __shared__ _Float16 As[BM * LDA];
  __shared__ _Float16 Bs[BN * LDA];

  const int tid  = threadIdx.x;
  const int lane = tid & 31;
  const int wv   = tid >> 5;
  const int wm   = (wv & 3) * 32;      // wave row offset in block tile
  const int wn   = (wv >> 2) * 32;     // wave col offset in block tile
  const int rowBase = blockIdx.y * BM;
  const int colBase = blockIdx.x * BN;
  const int mrow = lane & 15;
  const int hi   = lane >> 4;          // half-wave select

  v8f acc[2][2] = {};

  for (int kt = 0; kt < KDIM / KC; ++kt) {
    const int k0 = kt * KC;
    // stage A tile: BM*KC halves = 2048 16-byte units, 8 per thread
#pragma unroll
    for (int it = 0; it < (BM * KC / 8) / 256; ++it) {
      const int u = it * 256 + tid;
      const int r = u >> 4, c = (u & 15) * 8;
      async_cp16(Z + (size_t)(rowBase + r) * KDIM + k0 + c, &As[r * LDA + c]);
    }
    // stage B tile: BN*KC halves = 1024 units, 4 per thread
#pragma unroll
    for (int it = 0; it < (BN * KC / 8) / 256; ++it) {
      const int u = it * 256 + tid;
      const int r = u >> 4, c = (u & 15) * 8;
      async_cp16(Z + (size_t)(colBase + r) * KDIM + k0 + c, &Bs[r * LDA + c]);
    }
    async_wait_all();
    __syncthreads();

#pragma unroll
    for (int ks = 0; ks < KC / 32; ++ks) {
      const int k = ks * 32;
      v16h a[2], b[2];
      // A fragment (16x32 f16): lanes 0-15 hold K[0..7],K[16..23] of row m;
      // lanes 16-31 hold K[8..15],K[24..31].
#pragma unroll
      for (int i = 0; i < 2; ++i) {
        const _Float16* ap = &As[(wm + i * 16 + mrow) * LDA + k + hi * 8];
        v8h lo = *(const v8h*)ap;
        v8h hh = *(const v8h*)(ap + 16);
        a[i] = __builtin_shufflevector(lo, hh, 0,1,2,3,4,5,6,7,8,9,10,11,12,13,14,15);
      }
      // B fragment (32x16 f16): lane n holds K[0..7],K[16..23] (lanes 0-15) or
      // K[8..15],K[24..31] (lanes 16-31) of column n -> row (wn+n) of Z.
#pragma unroll
      for (int j = 0; j < 2; ++j) {
        const _Float16* bp = &Bs[(wn + j * 16 + mrow) * LDA + k + hi * 8];
        v8h lo = *(const v8h*)bp;
        v8h hh = *(const v8h*)(bp + 16);
        b[j] = __builtin_shufflevector(lo, hh, 0,1,2,3,4,5,6,7,8,9,10,11,12,13,14,15);
      }
#pragma unroll
      for (int i = 0; i < 2; ++i)
#pragma unroll
        for (int j = 0; j < 2; ++j)
          acc[i][j] = __builtin_amdgcn_wmma_f32_16x16x32_f16(
              false, a[i], false, b[j], (short)0, acc[i][j], false, false);
    }
    __syncthreads();
  }

  // Epilogue: C/D layout -> VGPR v, lanes 0-15: M=v; lanes 16-31: M=v+8; N=lane%16
  const int n  = lane & 15;
  const int mo = hi * 8;
#pragma unroll
  for (int i = 0; i < 2; ++i)
#pragma unroll
    for (int j = 0; j < 2; ++j)
#pragma unroll
      for (int v = 0; v < 8; ++v) {
        float val = acc[i][j][v];
        val = (val < THRESH) ? 0.0f : val;
        out[(size_t)(rowBase + wm + i * 16 + mo + v) * N_ROWS +
            (colBase + wn + j * 16 + n)] = val;
      }
}

// ---------------------------------------------------------------------------
// Kernel 3: per-row top-K mask, in place.  One block per row; row cached in
// LDS.  If nnz<=K (the common case: ~11 entries survive the 0.2 threshold on
// N(0,1) data) the row is already equal to the reference output.  Otherwise,
// two-level 1024-bin histogram finds the K-th largest; ties at the cutoff are
// kept in ascending-index order (stable top_k behavior).
// ---------------------------------------------------------------------------
#define NB   1024
#define VLO  0.1875f

__device__ __forceinline__ int bin1(float v) {
  int b = (int)((v - VLO) * 1024.0f);
  return b < 0 ? 0 : (b > NB - 1 ? NB - 1 : b);
}

__global__ __launch_bounds__(256) void topk_mask_kernel(
    float* __restrict__ sim, const int* __restrict__ Kp) {
  __shared__ float srow[N_ROWS];
  __shared__ int   hist[NB];
  __shared__ int   segc[256];
  __shared__ int   s_nnz, s_bin, s_cgt, s_sub, s_tieb;

  const int tid = threadIdx.x;
  float* rp = sim + (size_t)blockIdx.x * N_ROWS;

  if (tid == 0) s_nnz = 0;
  for (int i = tid; i < NB; i += 256) hist[i] = 0;
  __syncthreads();

  int ln = 0;
  for (int i = tid; i < N_ROWS; i += 256) {
    float v = rp[i];
    srow[i] = v;
    if (v > 0.0f) ln++;
  }
  atomicAdd(&s_nnz, ln);
  __syncthreads();

  const int K = *Kp;
  if (s_nnz <= K) return;  // nothing to mask off

  // level-1 histogram over surviving (positive) values
  for (int i = tid; i < N_ROWS; i += 256) {
    float v = srow[i];
    if (v > 0.0f) atomicAdd(&hist[bin1(v)], 1);
  }
  __syncthreads();
  if (tid == 0) {
    int cum = 0, b = NB - 1;
    for (; b > 0; --b) {
      if (cum + hist[b] >= K) break;
      cum += hist[b];
    }
    s_bin = b; s_cgt = cum;
  }
  __syncthreads();
  const int   b1    = s_bin;
  const int   cgt   = s_cgt;
  const float binLo = VLO + (float)b1 * (1.0f / 1024.0f);

  // level-2 histogram inside the cutoff bin
  for (int i = tid; i < NB; i += 256) hist[i] = 0;
  __syncthreads();
  for (int i = tid; i < N_ROWS; i += 256) {
    float v = srow[i];
    if (v > 0.0f && bin1(v) == b1) {
      int s = (int)((v - binLo) * (1024.0f * 1024.0f));
      s = s < 0 ? 0 : (s > NB - 1 ? NB - 1 : s);
      atomicAdd(&hist[s], 1);
    }
  }
  __syncthreads();
  if (tid == 0) {
    const int need = K - cgt;
    int cum = 0, s = NB - 1;
    for (; s > 0; --s) {
      if (cum + hist[s] >= need) break;
      cum += hist[s];
    }
    s_sub = s; s_tieb = need - cum;
  }
  __syncthreads();
  const int sub1 = s_sub;
  const int tieb = s_tieb;

  // index-ordered tie ranking: thread t owns contiguous segment [t*32, t*32+32)
  const int base = tid * 32;
  int myties = 0;
  for (int i = base; i < base + 32; ++i) {
    float v = srow[i];
    if (v <= 0.0f) continue;
    int b = bin1(v);
    if (b != b1) continue;
    int s = (int)((v - binLo) * (1024.0f * 1024.0f));
    s = s < 0 ? 0 : (s > NB - 1 ? NB - 1 : s);
    if (s == sub1) myties++;
  }
  segc[tid] = myties;
  __syncthreads();
  if (tid == 0) {
    int acc = 0;
    for (int t = 0; t < 256; ++t) { int c = segc[t]; segc[t] = acc; acc += c; }
  }
  __syncthreads();

  int rank = segc[tid];
  for (int i = base; i < base + 32; ++i) {
    float v = srow[i];
    if (v <= 0.0f) continue;
    int b = bin1(v);
    bool drop;
    if (b > b1) drop = false;
    else if (b < b1) drop = true;
    else {
      int s = (int)((v - binLo) * (1024.0f * 1024.0f));
      s = s < 0 ? 0 : (s > NB - 1 ? NB - 1 : s);
      if (s > sub1) drop = false;
      else if (s < sub1) drop = true;
      else { drop = (rank >= tieb); rank++; }
    }
    if (drop) rp[i] = 0.0f;
  }
}

// ---------------------------------------------------------------------------
extern "C" void kernel_launch(void* const* d_in, const int* in_sizes, int n_in,
                              void* d_out, int out_size, void* d_ws, size_t ws_size,
                              hipStream_t stream) {
  const float* x  = (const float*)d_in[0];   // [8192, 256] f32
  const float* w  = (const float*)d_in[1];   // [2, 256] f32
  const int*   Kp = (const int*)d_in[2];     // scalar K (=81)
  float*       out = (float*)d_out;          // [8192, 8192] f32
  _Float16*    Z   = (_Float16*)d_ws;        // [8192, 512] f16 (8 MB scratch)

  build_z_kernel<<<N_ROWS / 8, 256, 0, stream>>>(x, w, Z);

  dim3 g(N_ROWS / BN, N_ROWS / BM);          // (128, 64)
  gemm_sim_kernel<<<g, 256, 0, stream>>>(Z, out);

  topk_mask_kernel<<<N_ROWS, 256, 0, stream>>>(out, Kp);
}
